// PairMPNEncoder_12232066859192
// MI455X (gfx1250) — matrix operand order repaired and checked by reading, not compiled, not run on hardware
//
#include <hip/hip_runtime.h>
#include <hip/hip_bf16.h>
#include <stdint.h>

// ---------------------------------------------------------------------------
// PairMPNEncoder (chemprop D-MPNN) for MI455X / gfx1250, wave32 + WMMA bf16.
// GEMMs (the FLOP-dominant part) run through v_wmma_f32_16x16x32_bf16 with
// fp32 accumulation; gathers/scatter-pool are fp32 bandwidth kernels.
// ---------------------------------------------------------------------------

typedef __attribute__((ext_vector_type(16))) __bf16 v16bf;
typedef __attribute__((ext_vector_type(4)))  __bf16 v4bf;
typedef __attribute__((ext_vector_type(8)))  float  v8f;

#define HID     512
#define NATOMS  50000
#define NBONDS  100000
#define MAXNB   6
#define AFD     133
#define BFD     147
#define KB_PAD  160      // BFD padded to mult of 32
#define KO_PAD  672      // (AFD + HID)=645 padded to mult of 32
#define NMOLS   128
#define MB_PAD  100096   // NBONDS padded to mult of 128
#define MA_PAD  50048    // NATOMS padded to mult of 128

__device__ __forceinline__ __bf16 f2bf(float f) {
  uint32_t u = __builtin_bit_cast(uint32_t, f);
  u += 0x7FFFu + ((u >> 16) & 1u);            // round-to-nearest-even
  uint16_t h = (uint16_t)(u >> 16);
  return __builtin_bit_cast(__bf16, h);
}

// ---------------------------------------------------------------------------
// Utility kernels
// ---------------------------------------------------------------------------
__global__ void zero_f32_kernel(float* __restrict__ p, int n) {
  int t = blockIdx.x * blockDim.x + threadIdx.x;
  if (t < n) p[t] = 0.0f;
}

// f32 [rows, scols] row-major -> bf16 [rows, dcols] row-major, zero-pad cols
__global__ void cvt_pad_bf16_kernel(const float* __restrict__ src,
                                    __bf16* __restrict__ dst,
                                    int rows, int scols, int dcols) {
  int t = blockIdx.x * blockDim.x + threadIdx.x;
  int total = rows * dcols;
  if (t >= total) return;
  int r = t / dcols;
  int c = t - r * dcols;
  float v = (c < scols) ? src[(size_t)r * scols + c] : 0.0f;
  dst[(size_t)r * dcols + c] = f2bf(v);
}

// ---------------------------------------------------------------------------
// WMMA bf16 GEMM:  out[M,N] = epilogue( A[M,K](bf16) @ W[N,K](bf16)^T )
// Block: 256 threads (8 waves). Wave tile: 32x32 (2x2 wmma accumulators).
// Block tile: 128(M) x 64(N).  grid = (N/64, Mpad/128).
// A rows are padded to a multiple of 128 so fragment loads never go OOB;
// stores are guarded by row < M.
// Fragment layouts per CDNA5 ISA 7.12.2:
//   A 16x32 bf16: lanes 0-15 -> M=lane, K in {kb..kb+7, kb+16..kb+23}, kb=0;
//                 lanes 16-31 -> same M, kb=8.   (two 16B chunks per lane)
//   B 32x16 bf16: lanes 0-15 -> N=lane, K as A with kb=0; lanes 16-31 -> kb=8.
//                 B = W^T so lane reads two 16B chunks of W row n, as for A.
//   C/D 16x16 f32: vgpr r -> M=r (lanes 0-15) / M=r+8 (lanes 16-31), N=lane&15.
// ---------------------------------------------------------------------------
__device__ __forceinline__ void epilogue_store(
    v8f c, int m0, int n0, int M, int N,
    const float* __restrict__ addin, const float* __restrict__ bias,
    float* __restrict__ outRaw, float* __restrict__ outAct,
    int doRelu, int lane)
{
  const int col  = n0 + (lane & 15);
  const int rOff = (lane >> 4) << 3;
  const float bv = bias ? bias[col] : 0.0f;
#pragma unroll
  for (int r = 0; r < 8; ++r) {
    const int row = m0 + rOff + r;
    if (row < M) {
      const size_t idx = (size_t)row * N + col;
      float v = c[r] + bv;
      if (addin) v += addin[idx];
      if (outRaw) outRaw[idx] = v;
      if (outAct) outAct[idx] = doRelu ? fmaxf(v, 0.0f) : v;
    }
  }
}

__global__ __launch_bounds__(256) void gemm_bf16_wmma_kernel(
    const __bf16* __restrict__ A, const __bf16* __restrict__ W,
    const float* __restrict__ addin, const float* __restrict__ bias,
    float* __restrict__ outRaw, float* __restrict__ outAct,
    int M, int N, int K, int doRelu)
{
  const int lane  = threadIdx.x & 31;
  const int wave  = threadIdx.x >> 5;
  const int mBase = blockIdx.y * 128 + (wave >> 1) * 32;
  const int nBase = blockIdx.x * 64  + (wave & 1)  * 32;

  v8f acc00 = {}, acc01 = {}, acc10 = {}, acc11 = {};

  const int aRow0 = mBase + (lane & 15);
  const int aRow1 = aRow0 + 16;
  const int bRow0 = nBase + (lane & 15);
  const int bRow1 = bRow0 + 16;
  const int aK = (lane >> 4) << 3;   // 0 or 8
  const int bK = (lane >> 4) << 3;   // 0 or 8

  for (int k0 = 0; k0 < K; k0 += 32) {
    union { v16bf v; uint4 q[2]; } a0, a1, b0, b1;
    const uint4* pa0 = reinterpret_cast<const uint4*>(A + (size_t)aRow0 * K + k0 + aK);
    const uint4* pa1 = reinterpret_cast<const uint4*>(A + (size_t)aRow1 * K + k0 + aK);
    a0.q[0] = pa0[0]; a0.q[1] = pa0[2];   // K = kb..kb+7 , kb+16..kb+23
    a1.q[0] = pa1[0]; a1.q[1] = pa1[2];
    const uint4* pb0 = reinterpret_cast<const uint4*>(W + (size_t)bRow0 * K + k0 + bK);
    const uint4* pb1 = reinterpret_cast<const uint4*>(W + (size_t)bRow1 * K + k0 + bK);
    b0.q[0] = pb0[0]; b0.q[1] = pb0[2];   // K = kb..kb+7 , kb+16..kb+23
    b1.q[0] = pb1[0]; b1.q[1] = pb1[2];

    acc00 = __builtin_amdgcn_wmma_f32_16x16x32_bf16(false, a0.v, false, b0.v,
                                                    (short)0, acc00, false, false);
    acc01 = __builtin_amdgcn_wmma_f32_16x16x32_bf16(false, a0.v, false, b1.v,
                                                    (short)0, acc01, false, false);
    acc10 = __builtin_amdgcn_wmma_f32_16x16x32_bf16(false, a1.v, false, b0.v,
                                                    (short)0, acc10, false, false);
    acc11 = __builtin_amdgcn_wmma_f32_16x16x32_bf16(false, a1.v, false, b1.v,
                                                    (short)0, acc11, false, false);
  }

  epilogue_store(acc00, mBase,      nBase,      M, N, addin, bias, outRaw, outAct, doRelu, lane);
  epilogue_store(acc01, mBase,      nBase + 16, M, N, addin, bias, outRaw, outAct, doRelu, lane);
  epilogue_store(acc10, mBase + 16, nBase,      M, N, addin, bias, outRaw, outAct, doRelu, lane);
  epilogue_store(acc11, mBase + 16, nBase + 16, M, N, addin, bias, outRaw, outAct, doRelu, lane);
}

// ---------------------------------------------------------------------------
// a_message[a] = sum_j message[a2b[a][j]]   (float4-vectorized along H)
// ---------------------------------------------------------------------------
__global__ void gather_sum_kernel(const float* __restrict__ msg,
                                  const int* __restrict__ a2b,
                                  float* __restrict__ amsg, int natoms) {
  int t = blockIdx.x * blockDim.x + threadIdx.x;
  int total = natoms * (HID / 4);
  if (t >= total) return;
  int atom = t / (HID / 4);
  int h = (t - atom * (HID / 4)) * 4;
  float4 s = make_float4(0.f, 0.f, 0.f, 0.f);
#pragma unroll
  for (int j = 0; j < MAXNB; ++j) {
    int b = a2b[atom * MAXNB + j];
    const float4 v = *reinterpret_cast<const float4*>(msg + (size_t)b * HID + h);
    s.x += v.x; s.y += v.y; s.z += v.z; s.w += v.w;
  }
  *reinterpret_cast<float4*>(amsg + (size_t)atom * HID + h) = s;
}

// ---------------------------------------------------------------------------
// xh[b][h] = bf16( amsg[b2a[b]][h] - msg[b2revb[b]][h] )
// ---------------------------------------------------------------------------
__global__ void bond_update_kernel(const float* __restrict__ amsg,
                                   const float* __restrict__ msg,
                                   const int* __restrict__ b2a,
                                   const int* __restrict__ b2revb,
                                   __bf16* __restrict__ xh, int nbonds) {
  int t = blockIdx.x * blockDim.x + threadIdx.x;
  int total = nbonds * (HID / 4);
  if (t >= total) return;
  int b = t / (HID / 4);
  int h = (t - b * (HID / 4)) * 4;
  int a  = b2a[b];
  int rb = b2revb[b];
  const float4 va = *reinterpret_cast<const float4*>(amsg + (size_t)a  * HID + h);
  const float4 vm = *reinterpret_cast<const float4*>(msg  + (size_t)rb * HID + h);
  v4bf o;
  o[0] = f2bf(va.x - vm.x);
  o[1] = f2bf(va.y - vm.y);
  o[2] = f2bf(va.z - vm.z);
  o[3] = f2bf(va.w - vm.w);
  *reinterpret_cast<v4bf*>(xh + (size_t)b * HID + h) = o;
}

// ---------------------------------------------------------------------------
// ain[a] = bf16( concat(f_atoms[a], amsg[a]) zero-padded to KO_PAD )
// ---------------------------------------------------------------------------
__global__ void build_ain_kernel(const float* __restrict__ f_atoms,
                                 const float* __restrict__ amsg,
                                 __bf16* __restrict__ ain, int natoms) {
  int t = blockIdx.x * blockDim.x + threadIdx.x;
  int total = natoms * KO_PAD;
  if (t >= total) return;
  int a = t / KO_PAD;
  int c = t - a * KO_PAD;
  float v;
  if (c < AFD)            v = f_atoms[(size_t)a * AFD + c];
  else if (c < AFD + HID) v = amsg[(size_t)a * HID + (c - AFD)];
  else                    v = 0.0f;
  ain[(size_t)a * KO_PAD + c] = f2bf(v);
}

// ---------------------------------------------------------------------------
// Segment mean pool
// ---------------------------------------------------------------------------
__global__ void count_kernel(const int* __restrict__ mol_ids,
                             float* __restrict__ counts, int natoms) {
  int t = blockIdx.x * blockDim.x + threadIdx.x;
  if (t < natoms) atomicAdd(&counts[mol_ids[t]], 1.0f);
}

__global__ void pool_kernel(const float* __restrict__ ah,
                            const int* __restrict__ mol_ids,
                            float* __restrict__ sums, int natoms) {
  int t = blockIdx.x * blockDim.x + threadIdx.x;
  int total = natoms * (HID / 4);
  if (t >= total) return;
  int atom = t / (HID / 4);
  int h = (t - atom * (HID / 4)) * 4;
  int mol = mol_ids[atom];
  const float4 v = *reinterpret_cast<const float4*>(ah + (size_t)atom * HID + h);
  atomicAdd(&sums[(size_t)mol * HID + h + 0], v.x);
  atomicAdd(&sums[(size_t)mol * HID + h + 1], v.y);
  atomicAdd(&sums[(size_t)mol * HID + h + 2], v.z);
  atomicAdd(&sums[(size_t)mol * HID + h + 3], v.w);
}

__global__ void finalize_pool_kernel(float* __restrict__ sums,
                                     const float* __restrict__ counts, int nmols) {
  int t = blockIdx.x * blockDim.x + threadIdx.x;
  if (t >= nmols * HID) return;
  int m = t / HID;
  sums[t] /= fmaxf(counts[m], 1.0f);
}

// ---------------------------------------------------------------------------
// Host launcher
// ---------------------------------------------------------------------------
static inline int cdiv(long long a, long long b) { return (int)((a + b - 1) / b); }

extern "C" void kernel_launch(void* const* d_in, const int* in_sizes, int n_in,
                              void* d_out, int out_size, void* d_ws, size_t ws_size,
                              hipStream_t stream) {
  const float* f_atoms     = (const float*)d_in[0];
  const float* f_bonds     = (const float*)d_in[1];
  const float* ano_f_atoms = (const float*)d_in[2];
  const float* ano_f_bonds = (const float*)d_in[3];
  const float* Wi          = (const float*)d_in[4];
  const float* Wh          = (const float*)d_in[5];
  const float* Wo          = (const float*)d_in[6];
  const float* bo          = (const float*)d_in[7];
  const int*   a2b         = (const int*)d_in[8];
  const int*   b2a         = (const int*)d_in[9];
  const int*   b2revb      = (const int*)d_in[10];
  const int*   mol_ids     = (const int*)d_in[11];
  const int*   ano_a2b     = (const int*)d_in[12];
  const int*   ano_b2a     = (const int*)d_in[13];
  const int*   ano_b2revb  = (const int*)d_in[14];
  const int*   ano_mol_ids = (const int*)d_in[15];
  float* out = (float*)d_out;

  // ---------------- workspace partition (all offsets 256B-aligned) ----------
  char* ws = (char*)d_ws;
  float*  inp   = (float*)ws;                         ws += (size_t)MB_PAD * HID * 4;   // raw Wi output
  float*  msg   = (float*)ws;                         ws += (size_t)MB_PAD * HID * 4;   // relu'd message
  float*  amsg  = (float*)ws;                         ws += (size_t)MA_PAD * HID * 4;   // atom gather sum
  __bf16* xst   = (__bf16*)ws;                        ws += (size_t)MB_PAD * HID * 2;   // bf16 activation staging
  __bf16* WiB   = (__bf16*)ws;                        ws += (size_t)HID * KB_PAD * 2;
  __bf16* WhB   = (__bf16*)ws;                        ws += (size_t)HID * HID * 2;
  __bf16* WoB   = (__bf16*)ws;                        ws += (size_t)HID * KO_PAD * 2;
  float*  counts= (float*)ws;                         ws += 256 * 4;
  float*  atomh = inp;   // reuse: inp is dead after the last Wh GEMM of an encode

  const int T = 256;

  // zero outputs + counts (harness poisons once; we must init)
  zero_f32_kernel<<<cdiv(2 * NMOLS * HID, T), T, 0, stream>>>(out, 2 * NMOLS * HID);
  zero_f32_kernel<<<1, 256, 0, stream>>>(counts, 256);

  // weights -> bf16 (K zero-padded)
  cvt_pad_bf16_kernel<<<cdiv((long long)HID * KB_PAD, T), T, 0, stream>>>(Wi, WiB, HID, BFD, KB_PAD);
  cvt_pad_bf16_kernel<<<cdiv((long long)HID * HID,    T), T, 0, stream>>>(Wh, WhB, HID, HID, HID);
  cvt_pad_bf16_kernel<<<cdiv((long long)HID * KO_PAD, T), T, 0, stream>>>(Wo, WoB, HID, AFD + HID, KO_PAD);

  for (int e = 0; e < 2; ++e) {
    const float* fA   = e ? ano_f_atoms : f_atoms;
    const float* fB   = e ? ano_f_bonds : f_bonds;
    const int*   cA2B = e ? ano_a2b     : a2b;
    const int*   cB2A = e ? ano_b2a     : b2a;
    const int*   cREV = e ? ano_b2revb  : b2revb;
    const int*   cMID = e ? ano_mol_ids : mol_ids;
    float* sums = out + (size_t)e * NMOLS * HID;
    float* cnt  = counts + e * NMOLS;

    // 1) f_bonds -> bf16 staging [NBONDS, KB_PAD]
    cvt_pad_bf16_kernel<<<cdiv((long long)NBONDS * KB_PAD, T), T, 0, stream>>>(
        fB, xst, NBONDS, BFD, KB_PAD);

    // 2) inp = fB @ Wi^T ; msg = relu(inp)
    {
      dim3 grid(HID / 64, MB_PAD / 128);
      gemm_bf16_wmma_kernel<<<grid, 256, 0, stream>>>(
          xst, WiB, nullptr, nullptr, inp, msg, NBONDS, HID, KB_PAD, 1);
    }

    // 3) message passing: DEPTH-1 = 2 iterations
    for (int it = 0; it < 2; ++it) {
      gather_sum_kernel<<<cdiv((long long)NATOMS * (HID / 4), T), T, 0, stream>>>(
          msg, cA2B, amsg, NATOMS);
      bond_update_kernel<<<cdiv((long long)NBONDS * (HID / 4), T), T, 0, stream>>>(
          amsg, msg, cB2A, cREV, xst, NBONDS);
      dim3 grid(HID / 64, MB_PAD / 128);
      gemm_bf16_wmma_kernel<<<grid, 256, 0, stream>>>(
          xst, WhB, /*addin=*/inp, nullptr, nullptr, msg, NBONDS, HID, HID, 1);
    }

    // 4) final atom gather
    gather_sum_kernel<<<cdiv((long long)NATOMS * (HID / 4), T), T, 0, stream>>>(
        msg, cA2B, amsg, NATOMS);

    // 5) a_input = concat(f_atoms, a_message) -> bf16 [NATOMS, KO_PAD]
    build_ain_kernel<<<cdiv((long long)NATOMS * KO_PAD, T), T, 0, stream>>>(
        fA, amsg, xst, NATOMS);

    // 6) atom_hiddens = relu(a_input @ Wo^T + bo)
    {
      dim3 grid(HID / 64, MA_PAD / 128);
      gemm_bf16_wmma_kernel<<<grid, 256, 0, stream>>>(
          xst, WoB, nullptr, bo, nullptr, atomh, NATOMS, HID, KO_PAD, 1);
    }

    // 7) per-molecule mean pool
    count_kernel<<<cdiv(NATOMS, T), T, 0, stream>>>(cMID, cnt, NATOMS);
    pool_kernel<<<cdiv((long long)NATOMS * (HID / 4), T), T, 0, stream>>>(
        atomh, cMID, sums, NATOMS);
    finalize_pool_kernel<<<cdiv(NMOLS * HID, T), T, 0, stream>>>(sums, cnt, NMOLS);
  }
}
